// Head_24361054503517
// MI455X (gfx1250) — matrix operand order repaired and checked by reading, not compiled, not run on hardware
//
#include <hip/hip_runtime.h>
#include <hip/hip_bf16.h>

typedef __attribute__((ext_vector_type(16))) _Float16 v16h;
typedef __attribute__((ext_vector_type(8)))  _Float16 v8h;
typedef __attribute__((ext_vector_type(8)))  float    v8f;

#define BATCH 4
#define SEQ   2048
#define EMB   1024
#define HS    64
#define QT_PER_B 128      // SEQ/16
#define KSTEPS   32       // EMB/32

// Async global->LDS (CDNA5 ASYNCcnt engine) if this toolchain exposes it.
#if defined(__has_builtin)
#if __has_builtin(__builtin_amdgcn_global_load_async_to_lds_b128) && \
    __has_builtin(__builtin_amdgcn_s_wait_asynccnt)
#define USE_ASYNC_LDS 1
#endif
#endif
#ifndef USE_ASYNC_LDS
#define USE_ASYNC_LDS 0
#endif

// A-matrix (16x32 f16): lane L holds row M=L%16, half=L/16.
// element j -> K = j + 8*half + 8*(j>>3)   (half0: K {0..7,16..23})
__device__ __forceinline__ int kidxA(int j, int half) {
  return j + (half << 3) + ((j >> 3) << 3);
}
// B-matrix (32x16 f16): lane L holds col N=L%16, half=L/16.
// element j -> K = j + 8*half + 8*(j>>3)   (same as A)
__device__ __forceinline__ int kidxB(int j, int half) {
  return j + (half << 3) + ((j >> 3) << 3);
}

__device__ __forceinline__ v8f wmma_f16(v16h a, v16h b, v8f c) {
  return __builtin_amdgcn_wmma_f32_16x16x32_f16(false, a, false, b, (short)0, c,
                                                false, false);
}

#if USE_ASYNC_LDS
// Builtin signature (from hipcc diagnostic): param1 is
// 'int __vector(4) __device__ *' (AS1, generic vector kind, non-const);
// param2 is the AS3 counterpart; then two imm ints (offset, cpol).
typedef int v4i_ __attribute__((vector_size(16)));
typedef __attribute__((address_space(1))) v4i_* gas1_v4i;
typedef __attribute__((address_space(3))) v4i_* las3_v4i;

// Copy 2048 halfs (4KB) global -> LDS with 8 async b128 ops (16B/lane each).
__device__ __forceinline__ void async_copy_2k(const _Float16* gsrc,
                                              _Float16* ldst, int lane) {
  gas1_v4i g = (gas1_v4i)(gsrc + lane * 8);
  las3_v4i l = (las3_v4i)(ldst + lane * 8);
#pragma unroll
  for (int i = 0; i < 8; ++i)   // +i*32 v4i elements == +i*512 bytes
    __builtin_amdgcn_global_load_async_to_lds_b128(g + i * 32, l + i * 32, 0,
                                                   0);
}
#endif

// ---------------------------------------------------------------------------
// Kernel 1: pack Wq/Wk/Wv (1024x64 f32, row-major [k][n]) into f16 B-fragments.
// Layout: wpack[((mat*4 + nt)*32 + kstep)*512 + lane*16 + j]
// ---------------------------------------------------------------------------
__global__ __launch_bounds__(32) void pack_w_kernel(const float* __restrict__ Wq,
                                                    const float* __restrict__ Wk,
                                                    const float* __restrict__ Wv,
                                                    _Float16* __restrict__ wpack) {
  int blk  = blockIdx.x;          // 0..383 == mat*128 + nt*32 + kstep
  int mat  = blk >> 7;
  int rem  = blk & 127;
  int nt   = rem >> 5;
  int ks   = rem & 31;
  int lane = threadIdx.x, lanen = lane & 15, half = lane >> 4;
  const float* W = (mat == 0) ? Wq : ((mat == 1) ? Wk : Wv);
  _Float16* dst = wpack + (size_t)blk * 512 + lane * 16;
#pragma unroll
  for (int j = 0; j < 16; ++j) {
    int k = ks * 32 + kidxB(j, half);
    dst[j] = (_Float16)W[k * HS + nt * 16 + lanen];
  }
}

// ---------------------------------------------------------------------------
// Kernel 2: fused QKV projection + bias + RoPE + fragment packing.
// One wave handles 16 rows of x. Emits (for S^T = K*Q^T attention):
//   qp: Q^T B-fragments [((b*128+qt)*2+ks)*32+lane]*16  (lane=query, K=head dim,
//       Q pre-scaled by C^-0.5)
//   kp: K  A-fragments  [((b*128+kt)*2+ks)*32+lane]*16  (lane=key,   K=head dim)
//   vp: V  B-fragments  [((b*64+chunk)*4+nt)*32+lane]*16 (lane=head dim, K=key,
//       32-key chunks)
// ---------------------------------------------------------------------------
__global__ __launch_bounds__(32) void proj_rope_kernel(
    const float* __restrict__ x,
    const float* __restrict__ bq, const float* __restrict__ bk,
    const float* __restrict__ bv, const _Float16* __restrict__ wpack,
    _Float16* __restrict__ qp, _Float16* __restrict__ kp,
    _Float16* __restrict__ vp) {
  __shared__ _Float16 lds[16 * 64];

  int tile = blockIdx.x;          // b*128 + qt
  int b = tile >> 7, qt = tile & 127;
  int t0 = qt << 4;
  int lane = threadIdx.x, lanen = lane & 15, half = lane >> 4;

  v8f accQ[4] = {}, accK[4] = {}, accV[4] = {};

  const float* xrow = x + ((size_t)(b * SEQ + t0 + lanen)) * EMB;

  for (int ks = 0; ks < KSTEPS; ++ks) {
    if (ks + 1 < KSTEPS)
      __builtin_prefetch(xrow + (ks + 1) * 32, 0, 1);   // global_prefetch_b8
    // Build x A-fragment (f32 -> f16), 8 aligned float2 loads per lane.
    v16h a;
#pragma unroll
    for (int m = 0; m < 8; ++m) {
      int col = ks * 32 + kidxA(2 * m, half);
      float2 f = *(const float2*)(xrow + col);
      a[2 * m]     = (_Float16)f.x;
      a[2 * m + 1] = (_Float16)f.y;
    }
    const _Float16* wbase = wpack + (size_t)ks * 512 + lane * 16;
#pragma unroll
    for (int nt = 0; nt < 4; ++nt) {
      v16h bQ = *(const v16h*)(wbase + (size_t)(0 * 4 + nt) * 16384);
      accQ[nt] = wmma_f16(a, bQ, accQ[nt]);
      v16h bK = *(const v16h*)(wbase + (size_t)(1 * 4 + nt) * 16384);
      accK[nt] = wmma_f16(a, bK, accK[nt]);
      v16h bV = *(const v16h*)(wbase + (size_t)(2 * 4 + nt) * 16384);
      accV[nt] = wmma_f16(a, bV, accV[nt]);
    }
  }

  // Bias + RoPE (+ score prescale folded into Q). C/D layout:
  // element (r, lane) = row M = r + 8*half, col N = nt*16 + lane%16.
#pragma unroll
  for (int nt = 0; nt < 4; ++nt) {
    int n = nt * 16 + lanen;
    float bqv = bq[n], bkv = bk[n], bvv = bv[n];
    // freq = theta^(-2i/64), i = n/2
    float freq = __expf(-(2.0f * (float)(n >> 1) / 64.0f) * 9.210340371976184f);
#pragma unroll
    for (int r = 0; r < 8; ++r) {
      int t = t0 + r + 8 * half;
      float q = accQ[nt][r] + bqv;
      float k = accK[nt][r] + bkv;
      float s, c;
      __sincosf((float)t * freq, &s, &c);
      float qpart = __shfl_xor(q, 1, 32);   // pair partner (even<->odd head dim)
      float kpart = __shfl_xor(k, 1, 32);
      float sgn = (n & 1) ? 1.0f : -1.0f;   // even: ev*c - od*s ; odd: ev*s + od*c
      q = q * c + sgn * qpart * s;
      k = k * c + sgn * kpart * s;
      accQ[nt][r] = q * 0.03125f;           // * C^-0.5 = 1/32
      accK[nt][r] = k;
      accV[nt][r] += bvv;
    }
  }

  size_t qt_base = ((size_t)(b * QT_PER_B + qt)) * 2 * 512;

  // ---- Q: C-layout -> LDS [row][col] -> B-fragments of Q^T (N=query, K=hs) ----
#pragma unroll
  for (int nt = 0; nt < 4; ++nt)
#pragma unroll
    for (int r = 0; r < 8; ++r)
      lds[(r + 8 * half) * 64 + nt * 16 + lanen] = (_Float16)accQ[nt][r];
#pragma unroll
  for (int ks = 0; ks < 2; ++ks) {
    // kidxB = two contiguous 16B runs per lane
    v8h lo = *(const v8h*)&lds[lanen * 64 + ks * 32 + 8 * half];
    v8h hi = *(const v8h*)&lds[lanen * 64 + ks * 32 + 16 + 8 * half];
    v16h f;
#pragma unroll
    for (int j = 0; j < 8; ++j) { f[j] = lo[j]; f[j + 8] = hi[j]; }
    *(v16h*)(qp + qt_base + (size_t)ks * 512 + lane * 16) = f;
  }

  // ---- K: C-layout -> LDS -> A-fragments (M=key, K=hs) ----
#pragma unroll
  for (int nt = 0; nt < 4; ++nt)
#pragma unroll
    for (int r = 0; r < 8; ++r)
      lds[(r + 8 * half) * 64 + nt * 16 + lanen] = (_Float16)accK[nt][r];
#pragma unroll
  for (int ks = 0; ks < 2; ++ks) {
    // kidxA = two contiguous 16B runs per lane
    v8h lo = *(const v8h*)&lds[lanen * 64 + ks * 32 + 8 * half];
    v8h hi = *(const v8h*)&lds[lanen * 64 + ks * 32 + 16 + 8 * half];
    v16h f;
#pragma unroll
    for (int j = 0; j < 8; ++j) { f[j] = lo[j]; f[j + 8] = hi[j]; }
    *(v16h*)(kp + qt_base + (size_t)ks * 512 + lane * 16) = f;
  }

  // ---- V: C-layout -> LDS -> B-fragments (N=head dim, K=key). Chunk-key
  // parity*16+8*half+j goes to fragment half 'half', element 8*parity+j.
#pragma unroll
  for (int nt = 0; nt < 4; ++nt)
#pragma unroll
    for (int r = 0; r < 8; ++r)
      lds[(r + 8 * half) * 64 + nt * 16 + lanen] = (_Float16)accV[nt][r];
  int parity = qt & 1, chunk = qt >> 1;
  {
#pragma unroll
    for (int nt = 0; nt < 4; ++nt) {
      v8h f;
#pragma unroll
      for (int j = 0; j < 8; ++j)
        f[j] = lds[(8 * half + j) * 64 + nt * 16 + lanen];
      *(v8h*)(vp + (((size_t)(b * 64 + chunk) * 4 + nt) * 32 +
                    lane) * 16 + 8 * parity) = f;
    }
  }
}

// ---------------------------------------------------------------------------
// Kernel 3: causal flash attention, S^T form. One wave per 16-query tile,
// 32-key chunks. S^T = K*Q^T puts query on lanes (lane-local softmax stats)
// and makes the P A-fragment a pure register concatenation.
// K/V chunk blocks (4KB each, contiguous) are double-buffered into LDS via
// GLOBAL_LOAD_ASYNC_TO_LDS_B128 when the toolchain exposes the builtins.
// ---------------------------------------------------------------------------
__global__ __launch_bounds__(32) void attn_kernel(
    const _Float16* __restrict__ qp, const _Float16* __restrict__ kp,
    const _Float16* __restrict__ vp, const int* __restrict__ mask,
    float* __restrict__ out) {
#if USE_ASYNC_LDS
  __shared__ _Float16 stage[2][4096];   // [buf][K block 2048 | V block 2048]
#endif
  int tile = blockIdx.x;
  int b = tile >> 7, qt = tile & 127;
  int t0 = qt << 4;
  int lane = threadIdx.x, lanen = lane & 15, half = lane >> 4;

  size_t qbase = ((size_t)(b * QT_PER_B + qt)) * 2 * 512 + lane * 16;
  v16h qb0 = *(const v16h*)(qp + qbase);
  v16h qb1 = *(const v16h*)(qp + qbase + 512);

  v8f O[4] = {};
  float m = -__builtin_inff();
  float l = 0.0f;
  int tq = t0 + lanen;            // this lane's query position (both halves)

  int nchunks = (t0 + 47) >> 5;   // covers keys 0 .. t0+15

#if USE_ASYNC_LDS
  const _Float16* kcb = kp + (size_t)b * QT_PER_B * 1024;  // + ch*2048
  const _Float16* vcb = vp + (size_t)b * 64 * 2048;        // + ch*2048
  async_copy_2k(kcb, &stage[0][0], lane);                  // chunk 0 K
  async_copy_2k(vcb, &stage[0][2048], lane);               // chunk 0 V
#endif

  for (int ch = 0; ch < nchunks; ++ch) {
#if USE_ASYNC_LDS
    if (ch + 1 < nchunks) {
      async_copy_2k(kcb + (size_t)(ch + 1) * 2048, &stage[(ch + 1) & 1][0],
                    lane);
      async_copy_2k(vcb + (size_t)(ch + 1) * 2048, &stage[(ch + 1) & 1][2048],
                    lane);
      __builtin_amdgcn_s_wait_asynccnt(16);   // chunk ch landed; ch+1 in flight
    } else {
      __builtin_amdgcn_s_wait_asynccnt(0);
    }
    asm volatile("" ::: "memory");
    const _Float16* Lk = &stage[ch & 1][0];
    const _Float16* Lv = &stage[ch & 1][2048];
    v16h ka00 = *(const v16h*)(Lk + lane * 16);
    v16h ka01 = *(const v16h*)(Lk + 512 + lane * 16);
    v16h ka10 = *(const v16h*)(Lk + 1024 + lane * 16);
    v16h ka11 = *(const v16h*)(Lk + 1536 + lane * 16);
#else
    size_t kb = ((size_t)(b * QT_PER_B + ch * 2)) * 2 * 512 + lane * 16;
    v16h ka00 = *(const v16h*)(kp + kb);          // key tile 0, hs 0..31
    v16h ka01 = *(const v16h*)(kp + kb + 512);    // key tile 0, hs 32..63
    v16h ka10 = *(const v16h*)(kp + kb + 1024);   // key tile 1
    v16h ka11 = *(const v16h*)(kp + kb + 1536);
#endif

    v8f S0 = {}, S1 = {};                         // S^T: lane=query, vgpr=key
    S0 = wmma_f16(ka00, qb0, S0);
    S0 = wmma_f16(ka01, qb1, S0);
    S1 = wmma_f16(ka10, qb0, S1);
    S1 = wmma_f16(ka11, qb1, S1);

    // Per-lane: 16 scores for keys {ch*32+8*half+r} U {ch*32+16+8*half+r}.
    int kbase = ch * 32 + 8 * half;
    float s[16];
    float vmax = -__builtin_inff();
#pragma unroll
    for (int r = 0; r < 8; ++r) {
      float a0 = (kbase + r      <= tq) ? S0[r] : -__builtin_inff();
      float a1 = (kbase + 16 + r <= tq) ? S1[r] : -__builtin_inff();
      s[r] = a0;
      s[r + 8] = a1;
      vmax = fmaxf(vmax, fmaxf(a0, a1));
    }
    vmax = fmaxf(vmax, __shfl_xor(vmax, 16, 32));   // combine the two key-halves
    float mnew  = fmaxf(m, vmax);
    float scale = __expf(m - mnew);

    // P A-fragment is lane-local: element j <-> key kidxA(j,half), which is
    // exactly S0[j] (j<8) / S1[j-8] (j>=8) in this lane.
    v16h pa;
    float psum = 0.0f;
#pragma unroll
    for (int j = 0; j < 16; ++j) {
      float p = __expf(s[j] - mnew);
      psum += p;
      pa[j] = (_Float16)p;
    }
    psum += __shfl_xor(psum, 16, 32);
    l = l * scale + psum;
    m = mnew;

    // O is lane=hs, vgpr=query: broadcast per-query scale from query lanes.
#pragma unroll
    for (int r = 0; r < 8; ++r) {
      float sr = __shfl(scale, r + 8 * half, 32);
#pragma unroll
      for (int nt = 0; nt < 4; ++nt) O[nt][r] *= sr;
    }

#pragma unroll
    for (int nt = 0; nt < 4; ++nt) {
#if USE_ASYNC_LDS
      v16h vb = *(const v16h*)(Lv + nt * 512 + lane * 16);
#else
      v16h vb = *(const v16h*)(vp + (((size_t)(b * 64 + ch) * 4 + nt) * 32 +
                                     lane) * 16);
#endif
      O[nt] = wmma_f16(pa, vb, O[nt]);
    }
  }

#pragma unroll
  for (int r = 0; r < 8; ++r) {
    float lr = __shfl(l, r + 8 * half, 32);
    float inv = 1.0f / lr;
    int t = t0 + r + 8 * half;
    bool live = mask[b * SEQ + t] != 0;
#pragma unroll
    for (int nt = 0; nt < 4; ++nt) {
      float val = live ? O[nt][r] * inv : __builtin_nanf("");
      out[((size_t)(b * SEQ + t)) * HS + nt * 16 + lanen] = val;
    }
  }
}

// ---------------------------------------------------------------------------
extern "C" void kernel_launch(void* const* d_in, const int* in_sizes, int n_in,
                              void* d_out, int out_size, void* d_ws,
                              size_t ws_size, hipStream_t stream) {
  const float* x   = (const float*)d_in[0];
  const int*  mask = (const int*)d_in[1];
  const float* Wq  = (const float*)d_in[2];
  const float* bq  = (const float*)d_in[3];
  const float* Wk  = (const float*)d_in[4];
  const float* bk  = (const float*)d_in[5];
  const float* Wv  = (const float*)d_in[6];
  const float* bv  = (const float*)d_in[7];
  float* out = (float*)d_out;

  _Float16* wpack = (_Float16*)d_ws;                       // 196608 halfs
  _Float16* qp = wpack + (size_t)3 * 4 * 32 * 512;         // 524288 halfs
  _Float16* kp = qp + (size_t)BATCH * QT_PER_B * 2 * 512;
  _Float16* vp = kp + (size_t)BATCH * QT_PER_B * 2 * 512;  // 524288 halfs

  pack_w_kernel<<<dim3(384), dim3(32), 0, stream>>>(Wq, Wk, Wv, wpack);
  proj_rope_kernel<<<dim3(BATCH * QT_PER_B), dim3(32), 0, stream>>>(
      x, bq, bk, bv, wpack, qp, kp, vp);
  attn_kernel<<<dim3(BATCH * QT_PER_B), dim3(32), 0, stream>>>(qp, kp, vp, mask,
                                                               out);
}
